// HungarianMatcher_30288109372176
// MI455X (gfx1250) — matrix operand order repaired and checked by reading, not compiled, not run on hardware
//
#include <hip/hip_runtime.h>
#include <hip/hip_bf16.h>

// ---------------- problem constants (from reference setup_inputs) -----------
#define BB 16     // batch
#define QQ 8400   // queries / anchors
#define CC 80     // classes
#define GG 64     // ground truths per image
#define QT (QQ / 16)          // 525 q-tiles of 16 rows
#define CENTER_R (2.5f / 32.0f)

typedef __attribute__((ext_vector_type(16))) _Float16 v16h;
typedef __attribute__((ext_vector_type(8)))  float    v8f;
typedef __attribute__((ext_vector_type(4)))  int      v4i;

// gfx1250 async global->LDS path (guarded: falls back to plain loads)
#if defined(__has_builtin)
#if __has_builtin(__builtin_amdgcn_global_load_async_to_lds_b128) && \
    __has_builtin(__builtin_amdgcn_s_wait_asynccnt)
#define USE_ASYNC_LDS 1
typedef __attribute__((address_space(1))) v4i* as1_v4i_p;
typedef __attribute__((address_space(3))) v4i* as3_v4i_p;
#endif
#if __has_builtin(__builtin_amdgcn_rcpf)
#define FRCP(x) __builtin_amdgcn_rcpf(x)
#endif
#endif
#ifndef FRCP
#define FRCP(x) (1.0f / (x))
#endif

__device__ __forceinline__ _Float16 sigh(float x) {
    // sigmoid via v_exp + v_rcp (1 ulp rcp is plenty for a matching cost)
    return (_Float16)FRCP(1.0f + __expf(-x));
}
// constant-index element writes only (keeps the vector SROA-promotable)
#define SIG4(A, o, f)                                                   \
    { A[(o)] = sigh((f).x); A[(o) + 1] = sigh((f).y);                   \
      A[(o) + 2] = sigh((f).z); A[(o) + 3] = sigh((f).w); }

// ============================================================================
// Kernel 0: per-image one-hot(label) WMMA B-operands, precomputed once.
// Layout: bmat[((b*4 + gt4)*3 + ks)*32 + lane] : v16h (32B) per lane.
// 16-bit B 32x16 layout: lanes 0-15: N=lane, halves 0..7 -> K=0..7,
// halves 8..15 -> K=16..23 ; lanes 16-31: N=lane-16, K base +8.
// ============================================================================
__global__ __launch_bounds__(32) void onehot_kernel(
    const int* __restrict__ labels, unsigned int* __restrict__ bmat)
{
    const int b    = blockIdx.x;
    const int lane = threadIdx.x;
    const int n  = lane & 15;
    const int kb = (lane >> 4) * 8;
    for (int gt4 = 0; gt4 < 4; ++gt4) {
        const int lab = labels[b * GG + gt4 * 16 + n];
        for (int ks = 0; ks < 3; ++ks) {
            const int kofs = ks * 32;
            v16h Bm;
#pragma unroll
            for (int j = 0; j < 16; ++j)
                Bm[j] = (_Float16)((lab == kofs + kb + j + (j >= 8 ? 8 : 0)) ? 1.0f : 0.0f);
            size_t idx = (((size_t)b * 4 + gt4) * 3 + ks) * 32 + lane;
            *(v16h*)(bmat + idx * 8) = Bm;
        }
    }
}

// ============================================================================
// Kernel 1: fused cost + IoU.  One wave32 per 16-query tile.
//   p[q,g] = sigmoid(logits[q,label[g]]) computed as WMMA GEMM:
//   A = sigmoid(logits) (16x32 f16 tiles over C=80, zero-padded to 96)
//   B = precomputed one-hot (32x16 f16), D accumulates f32 16x16 (q x g).
// Then each lane finishes its 8 (q,g) cost entries from LDS-staged boxes.
// ============================================================================
__global__ __launch_bounds__(32) void cost_kernel(
    const float* __restrict__ logits,   // [B,Q,C]
    const float* __restrict__ pboxes,   // [B,Q,4] cxcywh
    const float* __restrict__ gboxes,   // [B,G,4] cxcywh
    const unsigned int* __restrict__ bmat,
    float* __restrict__ cost,           // [B,Q,G]
    float* __restrict__ ious)           // [B,Q,G]
{
    const int b    = blockIdx.x / QT;
    const int tq   = blockIdx.x % QT;
    const int qb   = tq * 16;
    const int lane = threadIdx.x;

    __shared__ float s_graw[GG][4]; // gt cxcywh raw (async-staged)
    __shared__ float s_gx[GG][4];   // gt xyxy
    __shared__ float s_gc[GG][2];   // gt center
    __shared__ float s_qb[16][4];   // query xyxy
    __shared__ float s_qa[16][2];   // anchor center (cx,cy)
    __shared__ float s_fg[16];      // fg_mask per query row

    // ---- stage GT boxes into LDS (gfx1250 async path when available) ----
#if USE_ASYNC_LDS
    for (int t = lane; t < GG; t += 32) {
        v4i* gsrc = (v4i*)(gboxes + ((size_t)b * GG + t) * 4);
        v4i* ldst = (v4i*)&s_graw[t][0];
        __builtin_amdgcn_global_load_async_to_lds_b128(
            (as1_v4i_p)gsrc, (as3_v4i_p)ldst, 0, 0);
    }
    __builtin_amdgcn_s_wait_asynccnt(0);
#else
    for (int t = lane; t < GG; t += 32) {
        const float4 gp = *(const float4*)(gboxes + ((size_t)b * GG + t) * 4);
        s_graw[t][0] = gp.x; s_graw[t][1] = gp.y;
        s_graw[t][2] = gp.z; s_graw[t][3] = gp.w;
    }
#endif
    // ---- stage this tile's 16 query boxes ----
    if (lane < 16) {
        const float4 qp = *(const float4*)(pboxes + ((size_t)b * QQ + qb + lane) * 4);
        s_qa[lane][0] = qp.x; s_qa[lane][1] = qp.y;
        s_qb[lane][0] = qp.x - 0.5f * qp.z; s_qb[lane][1] = qp.y - 0.5f * qp.w;
        s_qb[lane][2] = qp.x + 0.5f * qp.z; s_qb[lane][3] = qp.y + 0.5f * qp.w;
    }
    __syncthreads();

    // derive gt xyxy / centers
    for (int t = lane; t < GG; t += 32) {
        float cx = s_graw[t][0], cy = s_graw[t][1];
        float w  = s_graw[t][2], h  = s_graw[t][3];
        s_gc[t][0] = cx; s_gc[t][1] = cy;
        s_gx[t][0] = cx - 0.5f * w; s_gx[t][1] = cy - 0.5f * h;
        s_gx[t][2] = cx + 0.5f * w; s_gx[t][3] = cy + 0.5f * h;
    }
    __syncthreads();

    // ---- fg_mask per query row (any over 64 GTs) ----
    if (lane < 16) {
        float ax = s_qa[lane][0], ay = s_qa[lane][1];
        bool any_b = false, any_c = false;
        for (int g = 0; g < GG; ++g) {
            any_b |= (ax > s_gx[g][0]) & (ax < s_gx[g][2]) &
                     (ay > s_gx[g][1]) & (ay < s_gx[g][3]);
            any_c |= (ax > s_gc[g][0] - CENTER_R) & (ax < s_gc[g][0] + CENTER_R) &
                     (ay > s_gc[g][1] - CENTER_R) & (ay < s_gc[g][1] + CENTER_R);
        }
        s_fg[lane] = (any_b || any_c) ? 1.0f : 0.0f;
    }
    __syncthreads();

    // ---- build A = sigmoid(logits) tiles, f16, K=80 padded to 3x32 ----
    // 16-bit A 16x32 layout: lanes 0-15: M=lane, halves 0..7 -> K=0..7,
    // halves 8..15 -> K=16..23 ; lanes 16-31: M=lane-16, K base +8.
    const int m  = lane & 15;
    const int hk = (lane >> 4) * 8;
    const float* lrow = logits + ((size_t)b * QQ + qb + m) * CC;
    // gfx1250 prefetch of the next tile's logit rows (global_prefetch_b8)
    __builtin_prefetch(lrow + (size_t)16 * CC, 0, 0);

    const float* r = lrow + hk;   // 16B-aligned (rows are 320B)
    v16h A0, A1, A2;
    SIG4(A0, 0,  *(const float4*)(r + 0));
    SIG4(A0, 4,  *(const float4*)(r + 4));
    SIG4(A0, 8,  *(const float4*)(r + 16));
    SIG4(A0, 12, *(const float4*)(r + 20));
    SIG4(A1, 0,  *(const float4*)(r + 32));
    SIG4(A1, 4,  *(const float4*)(r + 36));
    SIG4(A1, 8,  *(const float4*)(r + 48));
    SIG4(A1, 12, *(const float4*)(r + 52));
    SIG4(A2, 0,  *(const float4*)(r + 64));
    SIG4(A2, 4,  *(const float4*)(r + 68));
#pragma unroll
    for (int j = 8; j < 16; ++j) A2[j] = (_Float16)0.0f;   // K=80..95 pad

    // ---- 4 g-tiles of 16 columns: precomputed one-hot B, 3 WMMA k-steps ----
    // Unrolled: 4 independent accumulator chains -> WMMA/VALU co-execution.
    const int n = lane & 15;
    const v16h* bmp = (const v16h*)bmat + (size_t)b * 4 * 3 * 32 + lane;
#pragma unroll
    for (int gt4 = 0; gt4 < 4; ++gt4) {
        const int gb = gt4 * 16;
        const v16h B0 = bmp[(gt4 * 3 + 0) * 32];
        const v16h B1 = bmp[(gt4 * 3 + 1) * 32];
        const v16h B2 = bmp[(gt4 * 3 + 2) * 32];
        v8f acc = {};
        acc = __builtin_amdgcn_wmma_f32_16x16x32_f16(
            false, A0, false, B0, (short)0, acc, false, false);
        acc = __builtin_amdgcn_wmma_f32_16x16x32_f16(
            false, A1, false, B1, (short)0, acc, false, false);
        acc = __builtin_amdgcn_wmma_f32_16x16x32_f16(
            false, A2, false, B2, (short)0, acc, false, false);

        // acc[i] -> (q = qb + i + 8*(lane>>4), g = gb + n)
        const int qhi = (lane >> 4) << 3;
        const int g   = gb + n;
        const float gx1 = s_gx[g][0], gy1 = s_gx[g][1];
        const float gx2 = s_gx[g][2], gy2 = s_gx[g][3];
        const float gcx = s_gc[g][0], gcy = s_gc[g][1];
        const float area_b = (gx2 - gx1) * (gy2 - gy1);
#pragma unroll
        for (int i = 0; i < 8; ++i) {
            const int qi = qhi + i;
            float p = fminf(fmaxf((float)acc[i], 0.0f), 1.0f);

            // focal-style class cost
            float neg = 0.75f * p * p * (-__logf(1.0f - p + 1e-8f));
            float omp = 1.0f - p;
            float pos = 0.25f * omp * omp * (-__logf(p + 1e-8f));
            float cost_class = pos - neg;

            // IoU / GIoU (v_rcp instead of exact division: no spill pressure)
            float ax1 = s_qb[qi][0], ay1 = s_qb[qi][1];
            float ax2 = s_qb[qi][2], ay2 = s_qb[qi][3];
            float area_a = (ax2 - ax1) * (ay2 - ay1);
            float iw = fminf(ax2, gx2) - fmaxf(ax1, gx1);
            float ih = fminf(ay2, gy2) - fmaxf(ay1, gy1);
            iw = fmaxf(iw, 0.0f); ih = fmaxf(ih, 0.0f);
            float inter = iw * ih;
            float uni   = area_a + area_b - inter;
            float iou   = inter * FRCP(uni);
            float ew = fmaxf(ax2, gx2) - fminf(ax1, gx1);
            float eh = fmaxf(ay2, gy2) - fminf(ay1, gy1);
            ew = fmaxf(ew, 0.0f); eh = fmaxf(eh, 0.0f);
            float enc  = ew * eh;
            float giou = iou - (enc - uni) * FRCP(enc);

            // center-prior mask
            float ax = s_qa[qi][0], ay = s_qa[qi][1];
            bool in_box = (ax > gx1) & (ax < gx2) & (ay > gy1) & (ay < gy2);
            bool in_ctr = (ax > gcx - CENTER_R) & (ax < gcx + CENTER_R) &
                          (ay > gcy - CENTER_R) & (ay < gcy + CENTER_R);
            float iibc = (in_box && in_ctr) ? 1.0f : 0.0f;

            float c = cost_class - 3.0f * giou
                    + 100.0f * (1.0f - iibc)
                    + 10000.0f * (1.0f - s_fg[qi]);

            size_t idx = ((size_t)b * QQ + qb + qi) * GG + g;
            cost[idx] = c;
            ious[idx] = iou;
        }
    }
}

// ============================================================================
// Kernel 2: per (b,g) column: top-10 IoUs -> dyn_k, then dyn_k-th smallest
// cost -> matching threshold. 256 threads, LDS tree merge of sorted-10 lists.
// ============================================================================
__device__ __forceinline__ void insert_desc(float (&a)[10], float v) {
    if (v <= a[9]) return;
    int i = 9;
    while (i > 0 && a[i - 1] < v) { a[i] = a[i - 1]; --i; }
    a[i] = v;
}
__device__ __forceinline__ void insert_asc(float (&a)[10], float v) {
    if (v >= a[9]) return;
    int i = 9;
    while (i > 0 && a[i - 1] > v) { a[i] = a[i - 1]; --i; }
    a[i] = v;
}
__device__ __forceinline__ void merge10(float* dst, const float* a,
                                        const float* b, bool desc) {
    float tmp[10];
    int i = 0, j = 0;
    for (int k = 0; k < 10; ++k) {
        bool takeA = desc ? (a[i] >= b[j]) : (a[i] <= b[j]);
        tmp[k] = takeA ? a[i++] : b[j++];
    }
    for (int k = 0; k < 10; ++k) dst[k] = tmp[k];
}

__global__ __launch_bounds__(256) void dynk_kernel(
    const float* __restrict__ ious, const float* __restrict__ cost,
    int* __restrict__ dynk, float* __restrict__ thresh)
{
    const int b = blockIdx.x / GG, g = blockIdx.x % GG;
    const int t = threadIdx.x;
    __shared__ float s_vals[256][10];
    __shared__ int   s_k;

    // phase 1: top-10 largest IoUs in this column
    float top[10];
    for (int i = 0; i < 10; ++i) top[i] = -__builtin_inff();
    for (int q = t; q < QQ; q += 256)
        insert_desc(top, ious[((size_t)b * QQ + q) * GG + g]);
    for (int i = 0; i < 10; ++i) s_vals[t][i] = top[i];
    __syncthreads();
    for (int s = 128; s >= 1; s >>= 1) {
        if (t < s) merge10(s_vals[t], s_vals[t], s_vals[t + s], true);
        __syncthreads();
    }
    if (t == 0) {
        float sum = 0.0f;
        for (int i = 0; i < 10; ++i) sum += s_vals[0][i];
        int k = (int)sum;
        if (k < 1)  k = 1;
        if (k > 10) k = 10;
        dynk[b * GG + g] = k;
        s_k = k;
    }
    __syncthreads();
    const int k = s_k;

    // phase 2: k-th smallest cost in this column = matching threshold
    float bot[10];
    for (int i = 0; i < 10; ++i) bot[i] = __builtin_inff();
    for (int q = t; q < QQ; q += 256)
        insert_asc(bot, cost[((size_t)b * QQ + q) * GG + g]);
    for (int i = 0; i < 10; ++i) s_vals[t][i] = bot[i];
    __syncthreads();
    for (int s = 128; s >= 1; s >>= 1) {
        if (t < s) merge10(s_vals[t], s_vals[t], s_vals[t + s], false);
        __syncthreads();
    }
    if (t == 0) thresh[b * GG + g] = s_vals[0][k - 1];
}

// ============================================================================
// Kernel 3: one block (1024 thr) per image. Matching bitmask per anchor is a
// u64 (G=64). Initial match + stale fix, then the while-loop (give every GT
// at least one anchor), then emit outputs.
// ============================================================================
__global__ __launch_bounds__(1024) void resolve_kernel(
    const float* __restrict__ cost, const float* __restrict__ thresh,
    unsigned long long* __restrict__ rowmask_ws,
    float* __restrict__ out_sel, float* __restrict__ out_assign,
    float* __restrict__ out_mqid)
{
    const int b = blockIdx.x;
    const int t = threadIdx.x;

    __shared__ float          s_th[GG];
    __shared__ int            s_colcnt[GG];
    __shared__ unsigned char  s_stale[QQ];      // 8400 B
    __shared__ unsigned short s_add[QQ];        // 16800 B: #(100000) addends/row
    __shared__ float          s_rv[1024];
    __shared__ int            s_ri[1024];
    __shared__ int            s_flag;

    if (t < GG) { s_th[t] = thresh[b * GG + t]; s_colcnt[t] = 0; }
    __syncthreads();

    const float* cb = cost + (size_t)b * QQ * GG;
    unsigned long long* rm = rowmask_ws + (size_t)b * QQ;

    // ---- phase 1: initial matching + stale conflict fix ----
    for (int q = t; q < QQ; q += 1024) {
        const float* cr = cb + (size_t)q * GG;
        unsigned long long mask = 0ull;
        float bc = __builtin_inff(); int bg = 0;
        for (int g = 0; g < GG; ++g) {
            float cv = cr[g];
            if (cv <= s_th[g]) mask |= 1ull << g;
            if (cv < bc) { bc = cv; bg = g; }
        }
        unsigned char st = (__popcll(mask) > 1) ? 1 : 0;
        if (st) mask = 1ull << bg;      // stale anchors -> argmin GT only
        s_stale[q] = st;
        s_add[q]   = 0;
        rm[q]      = mask;
        unsigned long long mm = mask;
        while (mm) { int g = __ffsll(mm) - 1; atomicAdd(&s_colcnt[g], 1); mm &= mm - 1; }
    }
    __syncthreads();

    // ---- phase 2: while any GT column unmatched ----
    for (int iter = 0; iter < GG; ++iter) {
        if (t == 0) s_flag = 0;
        __syncthreads();
        if (t < GG && s_colcnt[t] == 0) s_flag = 1;
        __syncthreads();
        if (!s_flag) break;

        // c2 = cost + 1e5 * matched_q  (accumulate addend counts per row)
        for (int q = t; q < QQ; q += 1024)
            if (rm[q]) s_add[q]++;
        __syncthreads();

        // per unmatched column: block argmin over q of c2
        for (int g = 0; g < GG; ++g) {
            if (s_colcnt[g] != 0) continue;   // uniform (shared) condition
            float bv = __builtin_inff(); int bi = 0;
            for (int q = t; q < QQ; q += 1024) {
                float v = cb[(size_t)q * GG + g] + 100000.0f * (float)s_add[q];
                if (v < bv) { bv = v; bi = q; }
            }
            s_rv[t] = bv; s_ri[t] = bi;
            __syncthreads();
            for (int s = 512; s > 0; s >>= 1) {
                if (t < s) {
                    if (s_rv[t + s] < s_rv[t] ||
                        (s_rv[t + s] == s_rv[t] && s_ri[t + s] < s_ri[t])) {
                        s_rv[t] = s_rv[t + s]; s_ri[t] = s_ri[t + s];
                    }
                }
                __syncthreads();
            }
            if (t == 0) rm[s_ri[0]] |= 1ull << g;
            __syncthreads();
        }

        // fix stale rows that picked up extra GTs (row argmin of cost)
        for (int q = t; q < QQ; q += 1024) {
            unsigned long long mk = rm[q];
            if (s_stale[q] && __popcll(mk) > 1) {
                const float* cr = cb + (size_t)q * GG;
                float bv = __builtin_inff(); int bg = 0;
                for (int g = 0; g < GG; ++g)
                    if (cr[g] < bv) { bv = cr[g]; bg = g; }
                rm[q] = 1ull << bg;
            }
        }
        __syncthreads();

        // recompute column counts
        if (t < GG) s_colcnt[t] = 0;
        __syncthreads();
        for (int q = t; q < QQ; q += 1024) {
            unsigned long long mm = rm[q];
            while (mm) { int g = __ffsll(mm) - 1; atomicAdd(&s_colcnt[g], 1); mm &= mm - 1; }
        }
        __syncthreads();
    }

    // ---- phase 3: outputs ----
    float* selp = out_sel    + (size_t)b * QQ;
    float* asp  = out_assign + (size_t)b * QQ;
    for (int q = t; q < QQ; q += 1024) {
        unsigned long long mk = rm[q];
        selp[q] = mk ? 1.0f : 0.0f;
        asp[q]  = (float)(mk ? (__ffsll(mk) - 1) : 0);
    }
    for (int g = 0; g < GG; ++g) {
        float bv = __builtin_inff(); int bi = 0;
        for (int q = t; q < QQ; q += 1024) {
            if ((rm[q] >> g) & 1ull) {
                float v = cb[(size_t)q * GG + g] + 100000.0f * (float)s_add[q];
                if (v < bv) { bv = v; bi = q; }
            }
        }
        s_rv[t] = bv; s_ri[t] = bi;
        __syncthreads();
        for (int s = 512; s > 0; s >>= 1) {
            if (t < s) {
                if (s_rv[t + s] < s_rv[t] ||
                    (s_rv[t + s] == s_rv[t] && s_ri[t + s] < s_ri[t])) {
                    s_rv[t] = s_rv[t + s]; s_ri[t] = s_ri[t + s];
                }
            }
            __syncthreads();
        }
        if (t == 0) out_mqid[(size_t)b * GG + g] = (float)s_ri[0];
        __syncthreads();
    }
}

// ============================================================================
extern "C" void kernel_launch(void* const* d_in, const int* in_sizes, int n_in,
                              void* d_out, int out_size, void* d_ws, size_t ws_size,
                              hipStream_t stream) {
    const float* logits = (const float*)d_in[0];   // [16,8400,80] f32
    const float* pboxes = (const float*)d_in[1];   // [16,8400,4]  f32
    const int*   labels = (const int*)  d_in[2];   // [16,64]      i32
    const float* gboxes = (const float*)d_in[3];   // [16,64,4]    f32

    const size_t BQG = (size_t)BB * QQ * GG;

    // workspace layout
    float* ws_cost  = (float*)d_ws;                        // BQG f32
    float* ws_ious  = ws_cost + BQG;                       // BQG f32
    float* ws_th    = ws_ious + BQG;                       // B*G f32
    int*   ws_dynk  = (int*)(ws_th + BB * GG);             // B*G i32
    unsigned long long* ws_rowmask =
        (unsigned long long*)(ws_dynk + BB * GG);          // B*Q u64
    unsigned int* ws_bmat =
        (unsigned int*)(ws_rowmask + (size_t)BB * QQ);     // B*4*3*32*8 u32

    // outputs (tuple order): selected[B,Q], gt_assign[B,Q], matched_qid[B,G]
    float* out_sel    = (float*)d_out;
    float* out_assign = out_sel + (size_t)BB * QQ;
    float* out_mqid   = out_assign + (size_t)BB * QQ;

    onehot_kernel<<<BB, 32, 0, stream>>>(labels, ws_bmat);
    cost_kernel<<<BB * QT, 32, 0, stream>>>(logits, pboxes, gboxes, ws_bmat,
                                            ws_cost, ws_ious);
    dynk_kernel<<<BB * GG, 256, 0, stream>>>(ws_ious, ws_cost, ws_dynk, ws_th);
    resolve_kernel<<<BB, 1024, 0, stream>>>(ws_cost, ws_th, ws_rowmask,
                                            out_sel, out_assign, out_mqid);
}
